// CausalSelfAttention_36481452212430
// MI455X (gfx1250) — matrix operand order repaired and checked by reading, not compiled, not run on hardware
//
#include <hip/hip_runtime.h>

// ---------------------------------------------------------------------------
// CausalSelfAttention forward for MI455X (gfx1250, wave32, WMMA).
//   K1: QKV GEMM (bf16 WMMA, f32 acc, register-staged pipelined LDS tiles)
//       + fused RoPE -> Q,K [B,nh,S,hd] bf16, V transposed [B,nh,hd,S] bf16.
//   K2: per-wave flash attention (16 queries/wave), bf16 WMMA for QK^T and PV,
//       f32 online softmax, P transposed C->A layout through LDS.
//   K3: output projection GEMM (bf16 WMMA) -> f32 d_out.
// ---------------------------------------------------------------------------

typedef __bf16 bf16_t;
typedef __attribute__((ext_vector_type(16))) __bf16 v16bf;
typedef __attribute__((ext_vector_type(8)))  float  v8f;

static constexpr int kHID   = 1024;
static constexpr int kNH    = 16;
static constexpr int kHD    = 64;
static constexpr int kSEQ   = 2048;
static constexpr int kB     = 2;
static constexpr int kBS    = kB * kSEQ;        // 4096 rows
static constexpr int kQKVN  = 3 * kHID;         // 3072
static constexpr size_t kHeadElems = (size_t)kB * kNH * kSEQ * kHD;  // 4 Mi elems

union Frag { uint4 q[2]; v16bf v; };

__device__ __forceinline__ v8f wmma_bf16(v16bf a, v16bf b, v8f c) {
  // (neg_a, A, neg_b, B, c_mod, C, reuse_a, reuse_b)
  return __builtin_amdgcn_wmma_f32_16x16x32_bf16(false, a, false, b,
                                                 (short)0, c, false, false);
}

__device__ __forceinline__ ushort4 pack_bf16x4(float4 a) {
  union { bf16_t b[4]; ushort4 u; } pk;
  pk.b[0] = (bf16_t)a.x; pk.b[1] = (bf16_t)a.y;
  pk.b[2] = (bf16_t)a.z; pk.b[3] = (bf16_t)a.w;
  return pk.u;
}

// ---------------------------------------------------------------------------
// Shared GEMM mainloop: C[128x128] tile of  A[M,K] * W[N,K]^T  (both row-major
// with K contiguous).  8 waves: wave w -> rows [(w&3)*32,+32), cols [(w>>2)*64,+64).
// Each wave holds acc[2][4] = 2 row-tiles x 4 col-tiles of 16x16 f32.
// Register-staged pipeline: global loads for tile i+1 issue while tile i's
// WMMAs run, so only LDS stores sit on the critical path between barriers.
// AT = float (convert to bf16 in LDS) or bf16_t (raw copy).
// ---------------------------------------------------------------------------
template <typename AT>
__device__ __forceinline__ void gemm_mainloop(const AT* __restrict__ A,
                                              const float* __restrict__ W,
                                              int K, int m0, int n0,
                                              bf16_t* sA, bf16_t* sW,
                                              v8f (&acc)[2][4]) {
  const int tid  = threadIdx.x;
  const int lane = tid & 31;
  const int wave = tid >> 5;
  const int wm   = wave & 3;    // 32-row group
  const int wn   = wave >> 2;   // 64-col group
  const int lm   = lane & 15;
  const int half = lane >> 4;

  constexpr bool kF32A = (sizeof(AT) == 4);
  float4 ra[4];   // staged A tile (f32 path)
  uint4  ra2[2];  // staged A tile (bf16 path)
  float4 rw[4];   // staged W tile (always f32 in global)

  auto load_tiles = [&](int kt) {
    if constexpr (kF32A) {
#pragma unroll
      for (int i = 0; i < 4; ++i) {
        int idx = tid + i * 256;             // 0..1023
        int r = idx >> 3, c4 = (idx & 7) * 4;
        ra[i] = *(const float4*)&A[(size_t)(m0 + r) * K + kt + c4];
      }
    } else {
#pragma unroll
      for (int i = 0; i < 2; ++i) {
        int idx = tid + i * 256;             // 0..511
        int r = idx >> 2, c8 = (idx & 3) * 8;
        ra2[i] = *(const uint4*)&A[(size_t)(m0 + r) * K + kt + c8];
      }
    }
#pragma unroll
    for (int i = 0; i < 4; ++i) {
      int idx = tid + i * 256;
      int r = idx >> 3, c4 = (idx & 7) * 4;
      rw[i] = *(const float4*)&W[(size_t)(n0 + r) * K + kt + c4];
    }
  };

  auto store_tiles = [&]() {
    if constexpr (kF32A) {
#pragma unroll
      for (int i = 0; i < 4; ++i) {
        int idx = tid + i * 256;
        int r = idx >> 3, c4 = (idx & 7) * 4;
        *(ushort4*)&sA[r * 40 + c4] = pack_bf16x4(ra[i]);
      }
    } else {
#pragma unroll
      for (int i = 0; i < 2; ++i) {
        int idx = tid + i * 256;
        int r = idx >> 2, c8 = (idx & 3) * 8;
        *(uint4*)&sA[r * 40 + c8] = ra2[i];
      }
    }
#pragma unroll
    for (int i = 0; i < 4; ++i) {
      int idx = tid + i * 256;
      int r = idx >> 3, c4 = (idx & 7) * 4;
      *(ushort4*)&sW[r * 40 + c4] = pack_bf16x4(rw[i]);
    }
  };

  load_tiles(0);  // prologue

  for (int kt = 0; kt < K; kt += 32) {
    store_tiles();
    __syncthreads();

    // Kick off next tile's global loads; they overlap the WMMAs below.
    if (kt + 32 < K) load_tiles(kt + 32);

    // ---- fragments ----
    Frag aF[2], bF[4];
#pragma unroll
    for (int mi = 0; mi < 2; ++mi) {
      const bf16_t* base = &sA[(wm * 32 + mi * 16 + lm) * 40];
      aF[mi].q[0] = *(const uint4*)(base + 8 * half);        // K = 0..7  (+8*half)
      aF[mi].q[1] = *(const uint4*)(base + 16 + 8 * half);   // K = 16..23(+8*half)
    }
#pragma unroll
    for (int ni = 0; ni < 4; ++ni) {
      const bf16_t* base = &sW[(wn * 64 + ni * 16 + lm) * 40];
      bF[ni].q[0] = *(const uint4*)(base + 8 * half);
      bF[ni].q[1] = *(const uint4*)(base + 16 + 8 * half);
    }
#pragma unroll
    for (int mi = 0; mi < 2; ++mi)
#pragma unroll
      for (int ni = 0; ni < 4; ++ni)
        acc[mi][ni] = wmma_bf16(aF[mi].v, bF[ni].v, acc[mi][ni]);
    __syncthreads();
  }
}

// ---------------------------------------------------------------------------
// Kernel 1: QKV projection + RoPE.  C = X[4096,1024] * Wqkv[3072,1024]^T.
// ---------------------------------------------------------------------------
__global__ __launch_bounds__(256) void qkv_rope_kernel(
    const float* __restrict__ X, const float* __restrict__ Wqkv,
    bf16_t* __restrict__ Q, bf16_t* __restrict__ Kmat, bf16_t* __restrict__ Vt) {
  __shared__ bf16_t sA[128 * 40];
  __shared__ bf16_t sW[128 * 40];
  v8f acc[2][4];
  v8f z = {};
#pragma unroll
  for (int i = 0; i < 2; ++i)
#pragma unroll
    for (int j = 0; j < 4; ++j) acc[i][j] = z;

  const int m0 = blockIdx.x * 128;
  const int n0 = blockIdx.y * 128;
  gemm_mainloop<float>(X, Wqkv, kHID, m0, n0, sA, sW, acc);

  const int tid = threadIdx.x, lane = tid & 31, wave = tid >> 5;
  const int wm = wave & 3, wn = wave >> 2, lm = lane & 15, half = lane >> 4;
  const int gc    = n0 + wn * 64;        // head strip base in [0,3072)
  const int which = gc >> 10;            // 0=Q 1=K 2=V
  const int head  = (gc >> 6) & 15;
  const float kfreq = -0.28782313662425572f;  // -ln(10000)/32

#pragma unroll
  for (int mi = 0; mi < 2; ++mi) {
#pragma unroll
    for (int v = 0; v < 8; ++v) {
      const int grow = m0 + wm * 32 + mi * 16 + v + 8 * half;  // row in [0,4096)
      const int b = grow >> 11;
      const int s = grow & 2047;
      if (which < 2) {
        bf16_t* dst = (which == 0 ? Q : Kmat) +
                      ((size_t)((b * 16 + head) * 2048 + s)) * 64;
#pragma unroll
        for (int ni = 0; ni < 2; ++ni) {
          const int j = ni * 16 + lm;            // 0..31
          float inv = __expf(kfreq * (float)j);
          float sn, cs;
          __sincosf((float)s * inv, &sn, &cs);
          float x1 = acc[mi][ni][v];
          float x2 = acc[mi][ni + 2][v];
          dst[j]      = (bf16_t)(x1 * cs - x2 * sn);
          dst[j + 32] = (bf16_t)(x2 * cs + x1 * sn);
        }
      } else {
#pragma unroll
        for (int ni = 0; ni < 4; ++ni) {
          const int d = ni * 16 + lm;
          Vt[((size_t)((b * 16 + head) * 64 + d)) * 2048 + s] =
              (bf16_t)acc[mi][ni][v];
        }
      }
    }
  }
}

// ---------------------------------------------------------------------------
// Kernel 2: flash attention, one wave per 16-query tile, 32-key steps.
// ---------------------------------------------------------------------------
__global__ __launch_bounds__(128) void attn_kernel(
    const bf16_t* __restrict__ Q, const bf16_t* __restrict__ Kmat,
    const bf16_t* __restrict__ Vt, bf16_t* __restrict__ AO) {
  __shared__ bf16_t sP[4][16 * 40];  // per-wave P staging (C-layout -> A-layout)

  const int tid  = threadIdx.x;
  const int lane = tid & 31;
  const int w    = tid >> 5;
  const int lm   = lane & 15;
  const int half = lane >> 4;

  const int task = blockIdx.x * 4 + w;          // 0..4095
  const int qblk = task & 127;
  const int head = (task >> 7) & 15;
  const int b    = task >> 11;
  const int q0   = qblk * 16;

  const bf16_t* Qh = Q    + ((size_t)(b * 16 + head)) * kSEQ * kHD;
  const bf16_t* Kh = Kmat + ((size_t)(b * 16 + head)) * kSEQ * kHD;
  const bf16_t* Vh = Vt   + ((size_t)(b * 16 + head)) * kHD * kSEQ;

  // Q as two A-fragments (d = 0..31, 32..63)
  Frag aQ[2];
#pragma unroll
  for (int kf = 0; kf < 2; ++kf) {
    const bf16_t* base = Qh + (size_t)(q0 + lm) * 64 + 32 * kf;
    aQ[kf].q[0] = *(const uint4*)(base + 8 * half);
    aQ[kf].q[1] = *(const uint4*)(base + 16 + 8 * half);
  }

  v8f Ot[4];
  v8f zz = {};
#pragma unroll
  for (int t = 0; t < 4; ++t) Ot[t] = zz;
  float mrun[8], lrun[8];
#pragma unroll
  for (int v = 0; v < 8; ++v) { mrun[v] = -1e30f; lrun[v] = 0.0f; }

  const int kend = q0 + 16;  // exclusive key bound (causal)
  for (int kb = 0; kb < kend; kb += 32) {
    // Prefetch next K/V tiles into cache (global_prefetch_b8).
    if (kb + 32 < kend) {
      __builtin_prefetch(Kh + (size_t)(kb + 32 + lm) * 64, 0, 3);
      __builtin_prefetch(Vh + (size_t)lm * kSEQ + kb + 32, 0, 3);
    }

    // ---- scores for 2 x 16-key tiles ----
    v8f sc[2];
#pragma unroll
    for (int sub = 0; sub < 2; ++sub) {
      const int k16 = kb + sub * 16;
      const bf16_t* kbase = Kh + (size_t)(k16 + lm) * 64;
      Frag bK0, bK1;
      bK0.q[0] = *(const uint4*)(kbase + 8 * half);
      bK0.q[1] = *(const uint4*)(kbase + 16 + 8 * half);
      bK1.q[0] = *(const uint4*)(kbase + 32 + 8 * half);
      bK1.q[1] = *(const uint4*)(kbase + 32 + 16 + 8 * half);
      v8f c = {};
      c = wmma_bf16(aQ[0].v, bK0.v, c);
      c = wmma_bf16(aQ[1].v, bK1.v, c);
      sc[sub] = c;
    }

    // ---- joint online softmax over the 32 keys ----
#pragma unroll
    for (int v = 0; v < 8; ++v) {
      const int qrow = q0 + v + 8 * half;
      float s0 = sc[0][v] * 0.125f;
      float s1 = sc[1][v] * 0.125f;
      if (kb + lm > qrow)      s0 = -1e30f;
      if (kb + 16 + lm > qrow) s1 = -1e30f;
      float mx = fmaxf(s0, s1);
      mx = fmaxf(mx, __shfl_xor(mx, 1));
      mx = fmaxf(mx, __shfl_xor(mx, 2));
      mx = fmaxf(mx, __shfl_xor(mx, 4));
      mx = fmaxf(mx, __shfl_xor(mx, 8));
      const float mnew  = fmaxf(mrun[v], mx);
      const float alpha = __expf(mrun[v] - mnew);
      const float p0 = __expf(s0 - mnew);
      const float p1 = __expf(s1 - mnew);
      float rs = p0 + p1;
      rs += __shfl_xor(rs, 1);
      rs += __shfl_xor(rs, 2);
      rs += __shfl_xor(rs, 4);
      rs += __shfl_xor(rs, 8);
      lrun[v] = lrun[v] * alpha + rs;
      mrun[v] = mnew;
#pragma unroll
      for (int t = 0; t < 4; ++t) Ot[t][v] *= alpha;
      const int m = v + 8 * half;
      sP[w][m * 40 + lm]      = (bf16_t)p0;
      sP[w][m * 40 + 16 + lm] = (bf16_t)p1;
    }

    // same-wave LDS write->read: DS ops are in-order within a wave; just
    // prevent compiler reordering.
    asm volatile("" ::: "memory");

    // ---- P (16x32) as A-fragment, PV accumulate ----
    Frag aP;
    {
      const bf16_t* base = &sP[w][lm * 40];
      aP.q[0] = *(const uint4*)(base + 8 * half);
      aP.q[1] = *(const uint4*)(base + 16 + 8 * half);
    }
#pragma unroll
    for (int t = 0; t < 4; ++t) {
      const bf16_t* vb = Vh + (size_t)(16 * t + lm) * kSEQ + kb;
      Frag bV;
      bV.q[0] = *(const uint4*)(vb + 8 * half);
      bV.q[1] = *(const uint4*)(vb + 16 + 8 * half);
      Ot[t] = wmma_bf16(aP.v, bV.v, Ot[t]);
    }
    asm volatile("" ::: "memory");
  }

  // ---- normalize + store (merge heads: AO[b,s,head*64+d], bf16) ----
#pragma unroll
  for (int v = 0; v < 8; ++v) {
    const float inv = 1.0f / lrun[v];
    const int s = q0 + v + 8 * half;
    const size_t rowoff = ((size_t)(b * kSEQ + s)) * kHID + head * 64;
#pragma unroll
    for (int t = 0; t < 4; ++t)
      AO[rowoff + t * 16 + lm] = (bf16_t)(Ot[t][v] * inv);
  }
}

// ---------------------------------------------------------------------------
// Kernel 3: output projection.  out = AO[4096,1024](bf16) * Wo[1024,1024]^T.
// ---------------------------------------------------------------------------
__global__ __launch_bounds__(256) void out_proj_kernel(
    const bf16_t* __restrict__ AO, const float* __restrict__ Wo,
    float* __restrict__ out) {
  __shared__ bf16_t sA[128 * 40];
  __shared__ bf16_t sW[128 * 40];
  v8f acc[2][4];
  v8f z = {};
#pragma unroll
  for (int i = 0; i < 2; ++i)
#pragma unroll
    for (int j = 0; j < 4; ++j) acc[i][j] = z;

  const int m0 = blockIdx.x * 128;
  const int n0 = blockIdx.y * 128;
  gemm_mainloop<bf16_t>(AO, Wo, kHID, m0, n0, sA, sW, acc);

  const int tid = threadIdx.x, lane = tid & 31, wave = tid >> 5;
  const int wm = wave & 3, wn = wave >> 2, lm = lane & 15, half = lane >> 4;
#pragma unroll
  for (int mi = 0; mi < 2; ++mi) {
#pragma unroll
    for (int v = 0; v < 8; ++v) {
      const int grow = m0 + wm * 32 + mi * 16 + v + 8 * half;
#pragma unroll
      for (int ni = 0; ni < 4; ++ni) {
        const int col = n0 + wn * 64 + ni * 16 + lm;
        out[(size_t)grow * kHID + col] = acc[mi][ni][v];
      }
    }
  }
}

// ---------------------------------------------------------------------------
extern "C" void kernel_launch(void* const* d_in, const int* in_sizes, int n_in,
                              void* d_out, int out_size, void* d_ws,
                              size_t ws_size, hipStream_t stream) {
  const float* X    = (const float*)d_in[0];   // [2,2048,1024] f32
  const float* Wqkv = (const float*)d_in[1];   // [3072,1024]   f32
  const float* Wo   = (const float*)d_in[2];   // [1024,1024]   f32
  float* out = (float*)d_out;                  // [2,2048,1024] f32

  // Workspace (bf16): Q | K | Vt | AO  == 4 * 8 MB = 32 MB.
  bf16_t* Q  = (bf16_t*)d_ws;
  bf16_t* K  = Q + kHeadElems;
  bf16_t* Vt = K + kHeadElems;
  bf16_t* AO = Vt + kHeadElems;

  qkv_rope_kernel<<<dim3(kBS / 128, kQKVN / 128), 256, 0, stream>>>(
      X, Wqkv, Q, K, Vt);
  attn_kernel<<<dim3((kB * kNH * (kSEQ / 16)) / 4), 128, 0, stream>>>(
      Q, K, Vt, AO);
  out_proj_kernel<<<dim3(kBS / 128, kHID / 128), 256, 0, stream>>>(
      AO, Wo, out);
}
